// SocialInteraction3_16716012716117
// MI455X (gfx1250) — matrix operand order repaired and checked by reading, not compiled, not run on hardware
//
#include <hip/hip_runtime.h>

typedef __attribute__((ext_vector_type(16))) _Float16 v16h;
typedef __attribute__((ext_vector_type(8)))  _Float16 v8h;
typedef __attribute__((ext_vector_type(8)))  float    v8f;
typedef __attribute__((ext_vector_type(4)))  int      v4i;

#define PN 8192
#define MD 64
#define NB 80        // 64 out cols + g + ones + 14 pad (5 WMMA N-tiles)
#define KSPLIT 8     // waves per block, each owns PN/KSPLIT of K
#define KCHUNK (PN / KSPLIT)

// ---------------- Kernel 1: row_i = h_i . w[:64] + b ; col_j = h_j . w[64:] ----
__global__ void k_rowcol(const float* __restrict__ h, const float* __restrict__ w,
                         const float* __restrict__ b, float* __restrict__ rowp,
                         float* __restrict__ col) {
  int j = blockIdx.x * blockDim.x + threadIdx.x;
  const float4* hr = (const float4*)(h + (size_t)j * MD);
  const float4* wa = (const float4*)(w);
  const float4* wb = (const float4*)(w + MD);
  float ra = 0.f, rb = 0.f;
#pragma unroll
  for (int q = 0; q < MD / 4; ++q) {
    float4 hv = hr[q], av = wa[q], bv = wb[q];
    ra += hv.x * av.x + hv.y * av.y + hv.z * av.z + hv.w * av.w;
    rb += hv.x * bv.x + hv.y * bv.y + hv.z * bv.z + hv.w * bv.w;
  }
  rowp[j] = ra + b[0];
  col[j]  = rb;
}

// ---------------- Kernel 2: colmax = max_j col_j ------------------------------
__global__ void k_colmax(const float* __restrict__ col, float* __restrict__ colmax) {
  __shared__ float sm[1024];
  float m = -1e30f;
  for (int j = threadIdx.x; j < PN; j += 1024) m = fmaxf(m, col[j]);
  sm[threadIdx.x] = m;
  __syncthreads();
  for (int s = 512; s > 0; s >>= 1) {
    if (threadIdx.x < (unsigned)s) sm[threadIdx.x] = fmaxf(sm[threadIdx.x], sm[threadIdx.x + s]);
    __syncthreads();
  }
  if (threadIdx.x == 0) *colmax = sm[0];
}

// ---------------- Kernel 3: GHT[n][k] (K-transposed B matrix, f16) + row_u ----
// GHT[n][k] = g_k*h[k][n] (n<64), g_k (n=64), 1 (n=65), 0 (n=66..79)
__global__ void k_build(const float* __restrict__ h, const float* __restrict__ col,
                        const float* __restrict__ colmax, const float* __restrict__ rowp,
                        _Float16* __restrict__ ght, float* __restrict__ row_u) {
  int j = blockIdx.x * blockDim.x + threadIdx.x;
  float cm = *colmax;
  float g  = __expf(col[j] - cm);
  const float* hr = h + (size_t)j * MD;
#pragma unroll 8
  for (int n = 0; n < MD; ++n) ght[(size_t)n * PN + j] = (_Float16)(g * hr[n]);
  ght[(size_t)64 * PN + j] = (_Float16)g;
  ght[(size_t)65 * PN + j] = (_Float16)1.0f;
#pragma unroll
  for (int n = 66; n < NB; ++n) ght[(size_t)n * PN + j] = (_Float16)0.0f;
  // weight of every unmasked entry in row i's softmax denominator
  row_u[j] = __expf(-1e-6f - rowp[j] - cm);
}

// ---------------- Kernel 4: masked GEMM via WMMA, K split across 8 waves ------
__device__ __forceinline__ unsigned pk2(int m0, int m1) {
  // nei values are {0,1}; (m0 + m1*2^16) * 0x3C00 == m0*0x3C00 | m1*0x3C000000
  // = packed f16x2 of {0.0|1.0, 0.0|1.0}.  (v_lshl_or + v_mul_lo)
  return (unsigned)(m0 | (m1 << 16)) * 0x3C00u;
}

__global__ __launch_bounds__(256)
void k_main(const int* __restrict__ nei, const _Float16* __restrict__ ght,
            const float* __restrict__ row_u, float* __restrict__ out) {
  __shared__ float part[KSPLIT * 16 * NB];          // 40 KB partial tiles

  const int lane = threadIdx.x & 31;
  const int lm   = lane & 15;
  const int half = lane >> 4;                       // 0 or 1 (lane half)
  const int wv   = threadIdx.x >> 5;                // K-chunk id, 0..7
  const int row0 = blockIdx.x * 16;                 // one 16-row tile per block
  const int kbeg = wv * KCHUNK;

  // A operand (16x32 f16): this lane covers mask row (row0+lm);
  // lane-half 0 holds K {0..7,16..23}, half 1 holds K {8..15,24..31}.
  const int* arow = nei + (size_t)(row0 + lm) * PN + kbeg + half * 8;

  v8f acc[5];
#pragma unroll
  for (int t = 0; t < 5; ++t) acc[t] = (v8f)0.0f;

  for (int k0 = 0; k0 < KCHUNK; k0 += 32) {
    const v4i* ap = (const v4i*)(arow + k0);
    // streaming, read-once: non-temporal loads + deep prefetch (1 KB/lane ahead)
    __builtin_prefetch(arow + k0 + 256, 0, 0);
    v4i a0 = __builtin_nontemporal_load(ap);        // K kh+0..3
    v4i a1 = __builtin_nontemporal_load(ap + 1);    // K kh+4..7
    v4i a2 = __builtin_nontemporal_load(ap + 4);    // K kh+16..19
    v4i a3 = __builtin_nontemporal_load(ap + 5);    // K kh+20..23
    union { unsigned u[8]; v16h h; } A;
    A.u[0] = pk2(a0[0], a0[1]); A.u[1] = pk2(a0[2], a0[3]);
    A.u[2] = pk2(a1[0], a1[1]); A.u[3] = pk2(a1[2], a1[3]);
    A.u[4] = pk2(a2[0], a2[1]); A.u[5] = pk2(a2[2], a2[3]);
    A.u[6] = pk2(a3[0], a3[1]); A.u[7] = pk2(a3[2], a3[3]);

    // B operand (32x16 f16): lane = column n, K {0..7,16..23} (half 0) / {8..15,24..31} (half 1)
    const _Float16* bcol = ght + (size_t)lm * PN + kbeg + k0 + half * 8;
#pragma unroll
    for (int t = 0; t < 5; ++t) {
      const v8h* bp = (const v8h*)(bcol + (size_t)t * 16 * PN);
      v8h lo = bp[0], hi = bp[2];
      v16h B = __builtin_shufflevector(lo, hi, 0,1,2,3,4,5,6,7,8,9,10,11,12,13,14,15);
      acc[t] = __builtin_amdgcn_wmma_f32_16x16x32_f16(
          false, A.h, false, B, (short)0, acc[t], false, false);
    }
  }

  // ---- dump per-wave partial 16x80 f32 tiles to LDS --------------------------
  // C/D layout: acc[t][r], lane L -> element (m = r + 8*half, n = t*16 + (L&15))
  float* myp = part + wv * (16 * NB);
#pragma unroll
  for (int t = 0; t < 5; ++t)
#pragma unroll
    for (int r = 0; r < 8; ++r)
      myp[(r + 8 * half) * NB + t * 16 + lm] = acc[t][r];
  __syncthreads();

  // ---- reduce 8 partials; write result into wave-0 slab ----------------------
  for (int e = threadIdx.x; e < 16 * NB; e += 256) {
    float s = part[e];
#pragma unroll
    for (int w = 1; w < KSPLIT; ++w) s += part[w * (16 * NB) + e];
    part[e] = s;                       // each element owned by exactly one thread
  }
  __syncthreads();

  // ---- finalize: out[i,c] = num[i,c] / (s_i + (P-cnt_i)*u_i) -----------------
  for (int e = threadIdx.x; e < 16 * MD; e += 256) {
    int r = e >> 6, c = e & 63;
    float s   = part[r * NB + 64];     // sum_j mask*g
    float cnt = part[r * NB + 65];     // # masked
    float u   = row_u[row0 + r];
    float inv = 1.0f / (s + ((float)PN - cnt) * u);
    out[(size_t)(row0 + r) * MD + c] = part[r * NB + c] * inv;
  }
}

// ---------------- Launch ------------------------------------------------------
extern "C" void kernel_launch(void* const* d_in, const int* in_sizes, int n_in,
                              void* d_out, int out_size, void* d_ws, size_t ws_size,
                              hipStream_t stream) {
  const float* h   = (const float*)d_in[0];   // hidden_state [8192,64]
  // d_in[1] = corr_index [8192,8192,2]  -- UNUSED by the reference; never read.
  const float* w   = (const float*)d_in[2];   // w_att [128]
  const float* b   = (const float*)d_in[3];   // b_att [1]
  const int*   nei = (const int*)d_in[4];     // nei_index [8192,8192]
  float* out = (float*)d_out;                 // [8192,64]

  char* ws = (char*)d_ws;
  float*    rowp   = (float*)(ws);            // 32 KB
  float*    col    = (float*)(ws + (32 << 10));
  float*    row_u  = (float*)(ws + (64 << 10));
  float*    colmax = (float*)(ws + (96 << 10));
  _Float16* ght    = (_Float16*)(ws + (128 << 10));  // 80*8192*2 = 1.25 MB

  k_rowcol<<<PN / 256, 256, 0, stream>>>(h, w, b, rowp, col);
  k_colmax<<<1, 1024, 0, stream>>>(col, colmax);
  k_build<<<PN / 256, 256, 0, stream>>>(h, col, colmax, rowp, ght, row_u);
  k_main<<<PN / 16, 256, 0, stream>>>(nei, ght, row_u, out);
}
